// NFM_84851373899957
// MI455X (gfx1250) — compile-verified, matrix operand order checked by reading, not run on hardware
//
#include <hip/hip_runtime.h>
#include <hip/hip_bf16.h>
#include <math.h>

typedef __attribute__((ext_vector_type(16))) _Float16 v16h;
typedef __attribute__((ext_vector_type(8)))  _Float16 v8h;
typedef __attribute__((ext_vector_type(4)))  _Float16 v4h;
typedef __attribute__((ext_vector_type(2)))  _Float16 v2h;
typedef __attribute__((ext_vector_type(8)))  float    v8f;
typedef __attribute__((ext_vector_type(4)))  unsigned int u32x4;
typedef __attribute__((ext_vector_type(8)))  int          i32x8;
typedef __attribute__((ext_vector_type(4)))  int          i32x4;

#define BATCH  16384
#define FIELDS 39
#define EDIM   16
#define HID    200
#define NT     13      // ceil(200/16)  -> N padded to 208
#define KTL    7       // ceil(200/32)  -> K padded to 224
#define NPAD   224
#define WLDS_HALVES ((1 + KTL + KTL) * NT * 512)   // 99840 halves = 199680 B
#define WLDS_QW     (WLDS_HALVES / 4)              // 24960 8-byte elements

// Single-instruction ReLU.
__device__ __forceinline__ float relu1(float x) {
    float r;
    asm("v_max_num_f32 %0, 0, %1" : "=v"(r) : "v"(x));
    return r;
}

// Packed ReLU'd f32 pair -> packed f16 (one v_cvt_pk_rtz_f16_f32).
__device__ __forceinline__ v2h relu_pk(float a, float b) {
    return __builtin_bit_cast(v2h, __builtin_amdgcn_cvt_pkrtz(relu1(a), relu1(b)));
}

// Per-lane K index inside a 32-deep k-tile for the f16 WMMA A/B operand layout.
__device__ __forceinline__ int klocal(int lane, int h) {
    int b = (h < 8) ? h : (h + 8);
    return b + ((lane & 16) ? 8 : 0);
}

// ---------------------------------------------------------------------------
// Repack fp32 W[Kreal, HID] (+ bias folded at row Kbias) into fp16
// fragment-major layout (fragment (kt,nt) = 512 contiguous halves; each lane's
// WMMA B operand is one contiguous 32B read). Padding rows/cols are zero.
// ---------------------------------------------------------------------------
__global__ void nfm_repack_kernel(const float* __restrict__ W,
                                  const float* __restrict__ bias,
                                  int Kreal, int Kbias, int ktiles,
                                  _Float16* __restrict__ dst) {
    int idx = blockIdx.x * blockDim.x + threadIdx.x;
    int total = ktiles * NT * 512;
    if (idx >= total) return;
    int frag   = idx >> 9;
    int within = idx & 511;
    int lane   = within >> 4;
    int h      = within & 15;
    int kt = frag / NT, nt = frag % NT;
    int n = nt * 16 + (lane & 15);
    int k = kt * 32 + klocal(lane, h);
    float val = 0.0f;
    if (n < HID) {
        if (k < Kreal)       val = W[k * HID + n];
        else if (k == Kbias) val = bias[n];
    }
    dst[idx] = (_Float16)val;
}

// ---------------------------------------------------------------------------
// Embedding gather + FM interaction + LR term, float4-vectorized:
// 4 lanes per batch row, one b128 load per lane per field (one coalesced 64B
// line per field), 8 rows in flight per wave, single 8B fp16 store per lane.
// ---------------------------------------------------------------------------
__global__ void nfm_gather_fm_kernel(const int*   __restrict__ ids,
                                     const float* __restrict__ vals,
                                     const float* __restrict__ w,
                                     const float* __restrict__ v,
                                     const float* __restrict__ gb,
                                     _Float16* __restrict__ fm,
                                     float* __restrict__ lr) {
    int tid = threadIdx.x;
    int q   = tid & 3;
    int row = tid >> 2;
    int b   = blockIdx.x * 64 + row;

    float4 xv = {0.f, 0.f, 0.f, 0.f};
    float4 x2 = {0.f, 0.f, 0.f, 0.f};
    float  lrp = 0.0f;
    for (int f = 0; f < FIELDS; ++f) {
        int    id  = ids[b * FIELDS + f];
        float  val = vals[b * FIELDS + f];
        float4 ve  = *(const float4*)(v + (size_t)id * EDIM + q * 4);
        xv.x += val * ve.x;  xv.y += val * ve.y;
        xv.z += val * ve.z;  xv.w += val * ve.w;
        float vv = val * val;
        x2.x += vv * ve.x * ve.x;  x2.y += vv * ve.y * ve.y;
        x2.z += vv * ve.z * ve.z;  x2.w += vv * ve.w * ve.w;
        if ((f & 3) == q) lrp += val * w[id];
    }
    lrp += __shfl_xor(lrp, 1, 32);
    lrp += __shfl_xor(lrp, 2, 32);

    v4h o;
    o[0] = (_Float16)(0.5f * (xv.x * xv.x - x2.x));
    o[1] = (_Float16)(0.5f * (xv.y * xv.y - x2.y));
    o[2] = (_Float16)(0.5f * (xv.z * xv.z - x2.z));
    o[3] = (_Float16)(0.5f * (xv.w * xv.w - x2.w));
    *(v4h*)(fm + (size_t)b * EDIM + q * 4) = o;
    if (q == 0) lr[b] = lrp + gb[0];
}

// ---------------------------------------------------------------------------
// Fused 3-layer MLP + row-sum + sigmoid. One wave = 16 batch rows.
// Packed weights staged into LDS once per block by the Tensor Data Mover
// (one wave issues, scalar-branch gated, TENSORcnt-synchronized); A operands
// in registers; activations staged per-wave in LDS; bias via constant-1.0
// activation column -> inline-zero accumulators.
// ---------------------------------------------------------------------------
__global__ void __launch_bounds__(256)
nfm_mlp_kernel(const _Float16* __restrict__ fm,
               const float*    __restrict__ lr,
               const _Float16* __restrict__ pk,   // pk0 | pk1 | pk2 contiguous
               float* __restrict__ out) {
    __shared__ __align__(16) _Float16 wlds[WLDS_HALVES];       // 199680 B
    __shared__ __align__(32) _Float16 act[8][16][NPAD];        //  57344 B

    const int tid   = threadIdx.x;
    const int wave  = tid >> 5;
    const int lane  = tid & 31;
    const int col   = lane & 15;
    const int rhalf = (lane & 16) ? 8 : 0;
    const int koff  = (lane & 16) ? 8 : 0;
    const int rowBase = (blockIdx.x * 8 + wave) * 16;

#if __has_builtin(__builtin_amdgcn_tensor_load_to_lds)
    // ---- TDM: DMA all packed weights (195KB) global -> LDS, one wave ----
    // readfirstlane makes the guard an SGPR compare -> scalar branch, so
    // waves 1..7 truly skip the issue (TDM ignores EXEC).
    if ((__builtin_amdgcn_readfirstlane(tid) >> 5) == 0) {
        unsigned lds_addr = (unsigned)(uintptr_t)(void*)&wlds[0];
        unsigned long long ga = (unsigned long long)(uintptr_t)pk;
        // D# group 0: count=1 | lds_addr | global_addr[56:0] | type=2
        u32x4 g0;
        g0[0] = 1u;
        g0[1] = lds_addr;
        g0[2] = (unsigned)(ga & 0xFFFFFFFFu);
        g0[3] = (unsigned)((ga >> 32) & 0x1FFFFFFu) | (2u << 30);
        // D# group 1: data_size=3 (8B); 1-row tile of WLDS_QW elements.
        i32x8 g1;
        g1[0] = (3 << 16);                         // data_size=8B
        g1[1] = (int)((unsigned)WLDS_QW << 16);    // tensor_dim0[15:0] @ bits 63:48
        g1[2] = (int)(1u << 16);                   // tensor_dim0 hi=0 | tensor_dim1=1
        g1[3] = (int)((unsigned)WLDS_QW << 16);    // tensor_dim1 hi=0 | tile_dim0
        g1[4] = 1;                                 // tile_dim1=1 | tile_dim2=0
        g1[5] = WLDS_QW;                           // tensor_dim0_stride lo32
        g1[6] = 0;                                 // stride hi | dim1_stride lo
        g1[7] = 0;
        i32x4 g2 = {0, 0, 0, 0};
        i32x4 g3 = {0, 0, 0, 0};
#if __clang_major__ <= 22
        __builtin_amdgcn_tensor_load_to_lds(g0, g1, g2, g3, 0);
#else
        i32x8 g4 = {0, 0, 0, 0, 0, 0, 0, 0};
        __builtin_amdgcn_tensor_load_to_lds(g0, g1, g2, g3, g4, 0);
#endif
        __builtin_amdgcn_s_wait_tensorcnt(0);
    }
#else
    {   // Fallback: block-cooperative 16B copy.
        const uint4* src = (const uint4*)pk;
        uint4*       dst = (uint4*)wlds;
        for (int i = tid; i < WLDS_HALVES / 8; i += 256) dst[i] = src[i];
    }
#endif
    // Activation padding: column 208 = 1.0 (bias lane), 209..223 = 0.
    for (int i = lane; i < 16 * 16; i += 32) {
        int r = i >> 4, cc = i & 15;
        act[wave][r][208 + cc] = (cc == 0) ? (_Float16)1.0f : (_Float16)0.0f;
    }
    __syncthreads();

    const _Float16* wl0 = wlds;
    const _Float16* wl1 = wlds + NT * 512;
    const _Float16* wl2 = wlds + (1 + KTL) * NT * 512;

    // ---------------- layer 0: fm[16x16] -> h0[16x208] ----------------
    v16h a0;
    {
        v8h lo = *(const v8h*)(fm + (size_t)(rowBase + col) * EDIM + koff);
        v8h hi = {(_Float16)0.0f,(_Float16)0.0f,(_Float16)0.0f,(_Float16)0.0f,
                  (_Float16)0.0f,(_Float16)0.0f,(_Float16)0.0f,(_Float16)0.0f};
        hi[0] = (lane < 16) ? (_Float16)1.0f : (_Float16)0.0f;   // bias input k=16
        a0 = __builtin_shufflevector(lo, hi, 0,1,2,3,4,5,6,7,8,9,10,11,12,13,14,15);
    }
    #pragma unroll 2
    for (int nt = 0; nt < NT; ++nt) {
        int n = nt * 16 + col;
        v8f c = {0.f, 0.f, 0.f, 0.f, 0.f, 0.f, 0.f, 0.f};
        v16h bf = *(const v16h*)(wl0 + (size_t)nt * 512 + lane * 16);
        c = __builtin_amdgcn_wmma_f32_16x16x32_f16(false, a0, false, bf,
                                                   (short)0, c, false, false);
        #pragma unroll
        for (int j = 0; j < 8; j += 2) {
            v2h p = relu_pk(c[j], c[j + 1]);
            act[wave][rhalf + j][n]     = p[0];
            act[wave][rhalf + j + 1][n] = p[1];
        }
    }

    // ---------------- layer 1: h0[16x208] -> h1[16x208] ----------------
    v16h A[KTL];
    #pragma unroll
    for (int kt = 0; kt < KTL; ++kt) {
        const _Float16* rp = &act[wave][col][kt * 32 + koff];
        v8h lo = *(const v8h*)rp;
        v8h hi = *(const v8h*)(rp + 16);
        A[kt] = __builtin_shufflevector(lo, hi, 0,1,2,3,4,5,6,7,8,9,10,11,12,13,14,15);
    }
    #pragma unroll 2
    for (int nt = 0; nt < NT; ++nt) {
        int n = nt * 16 + col;
        v8f c = {0.f, 0.f, 0.f, 0.f, 0.f, 0.f, 0.f, 0.f};
        #pragma unroll
        for (int kt = 0; kt < KTL; ++kt) {
            v16h bf = *(const v16h*)(wl1 + (size_t)(kt * NT + nt) * 512 + lane * 16);
            c = __builtin_amdgcn_wmma_f32_16x16x32_f16(false, A[kt], false, bf,
                                                       (short)0, c, false, false);
        }
        #pragma unroll
        for (int j = 0; j < 8; j += 2) {
            v2h p = relu_pk(c[j], c[j + 1]);
            act[wave][rhalf + j][n]     = p[0];
            act[wave][rhalf + j + 1][n] = p[1];
        }
    }

    // ---------------- layer 2: h1[16x208] -> rowsum(h2) ----------------
    #pragma unroll
    for (int kt = 0; kt < KTL; ++kt) {
        const _Float16* rp = &act[wave][col][kt * 32 + koff];
        v8h lo = *(const v8h*)rp;
        v8h hi = *(const v8h*)(rp + 16);
        A[kt] = __builtin_shufflevector(lo, hi, 0,1,2,3,4,5,6,7,8,9,10,11,12,13,14,15);
    }
    v8f rsum = {0.f, 0.f, 0.f, 0.f, 0.f, 0.f, 0.f, 0.f};
    #pragma unroll 2
    for (int nt = 0; nt < NT; ++nt) {
        v8f c = {0.f, 0.f, 0.f, 0.f, 0.f, 0.f, 0.f, 0.f};
        #pragma unroll
        for (int kt = 0; kt < KTL; ++kt) {
            v16h bf = *(const v16h*)(wl2 + (size_t)(kt * NT + nt) * 512 + lane * 16);
            c = __builtin_amdgcn_wmma_f32_16x16x32_f16(false, A[kt], false, bf,
                                                       (short)0, c, false, false);
        }
        #pragma unroll
        for (int j = 0; j < 8; ++j) rsum[j] += relu1(c[j]);
    }
    #pragma unroll
    for (int m = 1; m < 16; m <<= 1) {
        #pragma unroll
        for (int j = 0; j < 8; ++j) rsum[j] += __shfl_xor(rsum[j], m, 32);
    }
    if (col == 0) {
        #pragma unroll
        for (int j = 0; j < 8; ++j) {
            int r = rowBase + rhalf + j;
            float logit = lr[r] + rsum[j];
            out[r] = 1.0f / (1.0f + expf(-logit));
        }
    }
}

// ---------------------------------------------------------------------------
extern "C" void kernel_launch(void* const* d_in, const int* in_sizes, int n_in,
                              void* d_out, int out_size, void* d_ws, size_t ws_size,
                              hipStream_t stream) {
    (void)in_sizes; (void)n_in; (void)out_size; (void)ws_size;
    const int*   ids = (const int*)  d_in[0];
    const float* vals= (const float*)d_in[1];
    const float* w   = (const float*)d_in[2];
    const float* v   = (const float*)d_in[3];
    const float* gb  = (const float*)d_in[4];
    const float* w0  = (const float*)d_in[5];
    const float* b0  = (const float*)d_in[6];
    const float* w1  = (const float*)d_in[7];
    const float* b1  = (const float*)d_in[8];
    const float* w2  = (const float*)d_in[9];
    const float* b2  = (const float*)d_in[10];
    float* out = (float*)d_out;

    char* ws = (char*)d_ws;
    _Float16* fm  = (_Float16*)(ws);             // 16384*16*2  = 524288 B
    float*    lr  = (float*)   (ws + 524288);    // 16384*4     =  65536 B
    _Float16* pk0 = (_Float16*)(ws + 589824);    // 1*13*512*2  =  13312 B
    _Float16* pk1 = (_Float16*)(ws + 603136);    // 7*13*512*2  =  93184 B
    _Float16* pk2 = (_Float16*)(ws + 696320);    // 7*13*512*2  =  93184 B
    // pk0|pk1|pk2 form one contiguous 199680 B region starting at pk0.

    nfm_repack_kernel<<<(1  * NT * 512 + 255) / 256, 256, 0, stream>>>(w0, b0, EDIM, 16,  1,  pk0);
    nfm_repack_kernel<<<(KTL* NT * 512 + 255) / 256, 256, 0, stream>>>(w1, b1, HID, 208, KTL, pk1);
    nfm_repack_kernel<<<(KTL* NT * 512 + 255) / 256, 256, 0, stream>>>(w2, b2, HID, 208, KTL, pk2);

    nfm_gather_fm_kernel<<<BATCH / 64, 256, 0, stream>>>(ids, vals, w, v, gb, fm, lr);

    nfm_mlp_kernel<<<BATCH / 128, 256, 0, stream>>>(fm, lr, pk0, out);
}
